// Self_Attention_13211319402955
// MI455X (gfx1250) — matrix operand order repaired and checked by reading, not compiled, not run on hardware
//
#include <hip/hip_runtime.h>

// ---------------------------------------------------------------------------
// Self-attention (softmax over query axis) for B=8, C=256, H=W=64 on gfx1250.
// Flash-style two-pass attention; all matmuls via v_wmma_f32_16x16x32_f16.
//  k0: weights fp32 -> f16 (once).
//  k1: projections q,k,v with WMMA; input tile staged as f16 in LDS.
//  k2: per-output-column softmax stats (online max / log-sum-exp).
//  k3: cooperative 8-wave blocks over (batch, 64-m): waves produce the
//      exponentiated P tiles (QK WMMA + exp, computed exactly once) into LDS
//      in B-operand layout; all waves consume them for the PV WMMAs.
// Workspace: ~20.7 MB (qT,kT 2 MB f16; V 16 MB f16; stats 256 KB; wts 160 KB).
// ---------------------------------------------------------------------------

typedef __attribute__((ext_vector_type(16))) _Float16 v16h;
typedef __attribute__((ext_vector_type(8)))  _Float16 v8h;
typedef __attribute__((ext_vector_type(8)))  float    v8f;

constexpr int BATCH = 8;
constexpr int CH    = 256;    // channels
constexpr int C8    = 32;     // q/k channels
constexpr int NPIX  = 4096;   // H*W
constexpr int NT    = 256;    // NPIX / 16
constexpr int CHP2  = CH + 8; // padded LDS stride (halves) -> conflict-free

__device__ __forceinline__ v8f wmma_f16(v16h a, v16h b, v8f c) {
  // D = A(16x32 f16) * B(32x16 f16) + C(16x16 f32)
  return __builtin_amdgcn_wmma_f32_16x16x32_f16(
      /*neg_a=*/false, a, /*neg_b=*/false, b,
      /*c_mod=*/(short)0, c, /*reuse_a=*/false, /*reuse_b=*/false);
}

// A-operand (16-bit, 16x32, M=row, K=col): M = lane&15,
//   K = (lane>>4)*8 + h (h<8), 16 + (lane>>4)*8 + h-8 (h>=8).
// B-operand (16-bit, 32x16, K=row, N=col): N = lane&15, K as for A; stats/out use K = (lane>>4)*16 + h for A and B.
// C/D (f32 16x16): col N = lane&15, row M = (lane>>4)*8 + vgpr_index.

// ---------------------------------------------------------------------------
// Kernel 0: one-time weight conversion fp32 -> f16 (A-operands become pure
// b128 loads; removes per-block v_cvt streams from the projection GEMM).
// ---------------------------------------------------------------------------
__global__ __launch_bounds__(256) void wcvt_kernel(
    const float* __restrict__ wq, const float* __restrict__ wk,
    const float* __restrict__ wv,
    _Float16* __restrict__ wqh, _Float16* __restrict__ wkh,
    _Float16* __restrict__ wvh) {
  const int i = blockIdx.x * 256 + threadIdx.x;  // grid covers CH*CH = 65536
  wvh[i] = (_Float16)wv[i];
  if (i < C8 * CH) {
    wqh[i] = (_Float16)wq[i];
    wkh[i] = (_Float16)wk[i];
  }
}

// ---------------------------------------------------------------------------
// Kernel 1: projections q,k,v (1x1 conv). qT,kT stored [B][N][32] f16,
// V stored [B][C][N] f16. 4 waves/block; input tile staged transposed in LDS
// as f16 ([n][c], padded) so B-operands are 2x ds_load_b128, no cvt.
// ---------------------------------------------------------------------------
__global__ __launch_bounds__(128) void proj_kernel(
    const float* __restrict__ in,
    const _Float16* __restrict__ wqh, const float* __restrict__ bq,
    const _Float16* __restrict__ wkh, const float* __restrict__ bk,
    const _Float16* __restrict__ wvh, const float* __restrict__ bv,
    _Float16* __restrict__ qT, _Float16* __restrict__ kT,
    _Float16* __restrict__ vW) {
  __shared__ _Float16 sIn[16 * CHP2];  // [n][c] transposed f16, 8.25 KB

  const int b    = blockIdx.x / NT;
  const int nt   = blockIdx.x % NT;
  const int n0   = nt * 16;
  const int tid  = threadIdx.x;
  const int wave = tid >> 5;
  const int lane = tid & 31;

  // Stage input tile [256 c x 16 n] -> LDS transposed [16 n][264 c] f16.
  const float* src = in + (size_t)b * CH * NPIX;
  #pragma unroll 4
  for (int i = 0; i < 32; ++i) {
    int idx = i * 128 + tid;
    sIn[(idx & 15) * CHP2 + (idx >> 4)] =
        (_Float16)src[(size_t)(idx >> 4) * NPIX + n0 + (idx & 15)];
  }
  __syncthreads();

  const int ml = lane & 15;        // output column within tile / weight row
  const int rb = (lane >> 4) * 8;  // K-offset base for A, row base for C/D

  const _Float16* Wt[5]; const float* Bt[5]; int ob[5]; int kind[5];
  #pragma unroll
  for (int j = 0; j < 5; ++j) {
    int t = wave + j * 4;
    if (t < 2)      { Wt[j] = wqh; Bt[j] = bq; ob[j] = t * 16;       kind[j] = 0; }
    else if (t < 4) { Wt[j] = wkh; Bt[j] = bk; ob[j] = (t - 2) * 16; kind[j] = 1; }
    else            { Wt[j] = wvh; Bt[j] = bv; ob[j] = (t - 4) * 16; kind[j] = 2; }
  }

  v8f acc[5];
  #pragma unroll
  for (int j = 0; j < 5; ++j)
    acc[j] = *(const v8f*)(Bt[j] + ob[j] + rb);  // init with bias (32B aligned)

  for (int kc = 0; kc < CH; kc += 32) {
    // B operand: input [32 c x 16 n] from LDS; 2x b128 per lane, f16 direct.
    const _Float16* srow = &sIn[ml * CHP2 + kc + rb];
    v8h p0 = *(const v8h*)srow;
    v8h p1 = *(const v8h*)(srow + 16);
    v16h bm;
    #pragma unroll
    for (int h = 0; h < 8; ++h) { bm[h] = p0[h]; bm[h + 8] = p1[h]; }

    #pragma unroll
    for (int j = 0; j < 5; ++j) {
      const _Float16* wr = Wt[j] + (size_t)(ob[j] + ml) * CH + kc + rb;
      v8h w0 = *(const v8h*)wr;         // K = rb+0..7
      v8h w1 = *(const v8h*)(wr + 16);  // K = 16+rb+0..7
      v16h a;
      #pragma unroll
      for (int h = 0; h < 8; ++h) { a[h] = w0[h]; a[h + 8] = w1[h]; }
      acc[j] = wmma_f16(a, bm, acc[j]);
    }
  }

  const int nIdx = n0 + ml;
  #pragma unroll
  for (int j = 0; j < 5; ++j) {
    if (kind[j] == 2) {  // V stored [C][N]: 8 strided b16 stores
      #pragma unroll
      for (int h = 0; h < 8; ++h)
        vW[((size_t)b * CH + ob[j] + rb + h) * NPIX + nIdx] = (_Float16)acc[j][h];
    } else {             // qT/kT stored [N][32]: one contiguous 16B store
      _Float16* dst = (kind[j] == 0 ? qT : kT) +
                      ((size_t)b * NPIX + nIdx) * C8 + ob[j] + rb;
      v8h st;
      #pragma unroll
      for (int h = 0; h < 8; ++h) st[h] = (_Float16)acc[j][h];
      *(v8h*)dst = st;
    }
  }
}

__device__ __forceinline__ v16h load_qA(const _Float16* qbase, int n, int co) {
  const _Float16* qp = qbase + (size_t)n * C8 + 2 * co;
  v8h lo = *(const v8h*)qp;         // K = 2*co+0..7
  v8h hi = *(const v8h*)(qp + 8);   // K = 2*co+8..15
  v16h a;
  #pragma unroll
  for (int h = 0; h < 8; ++h) { a[h] = lo[h]; a[h + 8] = hi[h]; }
  return a;
}

// ---------------------------------------------------------------------------
// Kernel 2: per output-column softmax stats (online max / log-sum-exp).
// One wave per 16-m tile; S tile = single WMMA (K = C8 = 32).
// ---------------------------------------------------------------------------
__global__ __launch_bounds__(256) void stats_kernel(
    const _Float16* __restrict__ qT, const _Float16* __restrict__ kT,
    float* __restrict__ Mst, float* __restrict__ Lst) {
  const int wid  = blockIdx.x * 8 + (threadIdx.x >> 5);
  const int b    = wid / NT;
  const int m0   = (wid % NT) * 16;
  const int lane = threadIdx.x & 31;
  const int ml   = lane & 15;
  const int co   = (lane >> 4) * 8;

  const v16h bK = *(const v16h*)(kT + ((size_t)b * NPIX + m0 + ml) * C8 +
                                 (lane >> 4) * 16);
  const _Float16* qbase = qT + (size_t)b * NPIX * C8;

  float runM = -3.0e38f, runL = 0.0f;
  for (int n0 = 0; n0 < NPIX; n0 += 16) {
    v16h a = load_qA(qbase, n0 + ml, co);
    v8f z = {};
    v8f s = wmma_f16(a, bK, z);  // s[h] = score(n = n0 + rowbase + h, m)
    float cm = s[0];
    #pragma unroll
    for (int h = 1; h < 8; ++h) cm = fmaxf(cm, s[h]);
    float nm = fmaxf(runM, cm);
    float l  = runL * __expf(runM - nm);
    #pragma unroll
    for (int h = 0; h < 8; ++h) l += __expf(s[h] - nm);
    runM = nm; runL = l;
  }
  // Lane and lane^16 hold disjoint n subsets of the same column m: merge.
  float oM = __shfl_xor(runM, 16);
  float oL = __shfl_xor(runL, 16);
  float Mf = fmaxf(runM, oM);
  float Lf = runL * __expf(runM - Mf) + oL * __expf(oM - Mf);
  if (lane < 16) {
    Mst[(size_t)b * NPIX + m0 + ml] = Mf;
    Lst[(size_t)b * NPIX + m0 + ml] = Lf;
  }
}

// ---------------------------------------------------------------------------
// Kernel 3: O[c,m] = sum_n V[c,n] * exp(s(n,m)-M[m]) / L[m]; out = g*O + in.
// Block = 8 waves, one (batch, 64-m) block. Per 64-n iteration:
//   produce: wave w builds P for (32n subchunk w&1, 16m tile w>>1):
//            2 QK WMMAs + exp + cross-half shuffle repack -> LDS (B layout).
//   consume: each wave accumulates its 2 c-tiles against all 8 P subtiles:
//            16 PV WMMAs; V-tile loads amortized over 4 m-tiles.
// Scores are computed exactly once per (n,m) -- no redundancy.
// ---------------------------------------------------------------------------
__global__ __launch_bounds__(256) void out_kernel(
    const _Float16* __restrict__ qT, const _Float16* __restrict__ kT,
    const _Float16* __restrict__ vW,
    const float* __restrict__ Mst, const float* __restrict__ Lst,
    const float* __restrict__ gamma,
    const float* __restrict__ in, float* __restrict__ out) {
  __shared__ _Float16 pbuf[8 * 512];  // 8 subtiles x (32 lanes x 16 h) = 8 KB

  const int b    = blockIdx.x / (NT / 4);
  const int m0   = (blockIdx.x % (NT / 4)) * 64;
  const int tid  = threadIdx.x;
  const int w    = tid >> 5;
  const int lane = tid & 31;
  const int ml   = lane & 15;
  const int rb   = (lane >> 4) * 8;
  const bool lowHalf = lane < 16;

  // Producer role: n-subchunk s (0/1), m-tile mh (0..3).
  const int s     = w & 1;
  const int mh    = w >> 1;
  const int mProd = m0 + mh * 16;
  const v16h bK = *(const v16h*)(kT + ((size_t)b * NPIX + mProd + ml) * C8 +
                                 (lane >> 4) * 16);
  const float Mf = Mst[(size_t)b * NPIX + mProd + ml];
  _Float16* pslot = pbuf + (s * 4 + mh) * 512 + lane * 16;

  // Consumer role: c-tiles w and w+8.
  const _Float16* qbase = qT + (size_t)b * NPIX * C8;
  const _Float16* vb0 = vW + ((size_t)b * CH + w * 16 + ml) * NPIX;
  const _Float16* vb1 = vW + ((size_t)b * CH + (w + 8) * 16 + ml) * NPIX;

  v8f oacc[2][4] = {};  // [c-tile][m-tile]

  for (int nb = 0; nb < NPIX; nb += 64) {
    // ---- produce P for (subchunk s, m-tile mh) ----
    const int n0 = nb + s * 32;
    if (nb + 64 < NPIX) {  // uniform guard; hint next iteration into cache
      __builtin_prefetch(qbase + (size_t)(n0 + 64 + ml) * C8, 0, 3);
      __builtin_prefetch(vb0 + nb + 64 + rb, 0, 3);
      __builtin_prefetch(vb1 + nb + 64 + rb, 0, 3);
    }
    v16h aq0 = load_qA(qbase, n0 + ml, rb);
    v16h aq1 = load_qA(qbase, n0 + 16 + ml, rb);
    v8f z = {};
    v8f sA = wmma_f16(aq0, bK, z);
    v8f sB = wmma_f16(aq1, bK, z);

    float pA[8], pB[8];
    #pragma unroll
    for (int h = 0; h < 8; ++h) {
      pA[h] = __expf(sA[h] - Mf);
      pB[h] = __expf(sB[h] - Mf);
    }
    // Repack C/D layout into 32x16 B-operand layout (cross-half rows from the
    // partner lane, which shares the same output column m).
    v16h pm;
    #pragma unroll
    for (int h = 0; h < 8; ++h) {
      float xA = __shfl_xor(pA[h], 16);
      float xB = __shfl_xor(pB[h], 16);
      pm[h]     = (_Float16)(lowHalf ? pA[h] : xB);
      pm[h + 8] = (_Float16)(lowHalf ? xA : pB[h]);
    }
    *(v16h*)pslot = pm;  // per-lane 32B blob; consumers read the same shape
    __syncthreads();

    // ---- consume all 8 P subtiles for this wave's 2 c-tiles ----
    #pragma unroll
    for (int ss = 0; ss < 2; ++ss) {
      const int nn = nb + ss * 32 + (lane >> 4) * 16;
      v8h a0 = *(const v8h*)(vb0 + nn);
      v8h a1 = *(const v8h*)(vb0 + nn + 8);
      v8h b0 = *(const v8h*)(vb1 + nn);
      v8h b1 = *(const v8h*)(vb1 + nn + 8);
      v16h av0, av1;
      #pragma unroll
      for (int h = 0; h < 8; ++h) {
        av0[h] = a0[h]; av0[h + 8] = a1[h];
        av1[h] = b0[h]; av1[h + 8] = b1[h];
      }
      #pragma unroll
      for (int mm = 0; mm < 4; ++mm) {
        v16h pmat = *(const v16h*)(pbuf + (ss * 4 + mm) * 512 + lane * 16);
        oacc[0][mm] = wmma_f16(av0, pmat, oacc[0][mm]);
        oacc[1][mm] = wmma_f16(av1, pmat, oacc[1][mm]);
      }
    }
    __syncthreads();  // P fully consumed before next iteration overwrites it
  }

  const float g = gamma[0];
  float invL[4];
  #pragma unroll
  for (int mm = 0; mm < 4; ++mm)
    invL[mm] = 1.0f / Lst[(size_t)b * NPIX + m0 + mm * 16 + ml];
  #pragma unroll
  for (int cti = 0; cti < 2; ++cti) {
    const int ct = w + cti * 8;
    #pragma unroll
    for (int mm = 0; mm < 4; ++mm) {
      const size_t base =
          ((size_t)b * CH + ct * 16 + rb) * NPIX + m0 + mm * 16 + ml;
      #pragma unroll
      for (int h = 0; h < 8; ++h) {
        size_t off = base + (size_t)h * NPIX;
        out[off] = g * (oacc[cti][mm][h] * invL[mm]) + in[off];
      }
    }
  }
}

// ---------------------------------------------------------------------------
extern "C" void kernel_launch(void* const* d_in, const int* in_sizes, int n_in,
                              void* d_out, int out_size, void* d_ws,
                              size_t ws_size, hipStream_t stream) {
  const float* input = (const float*)d_in[0];
  const float* wq    = (const float*)d_in[1];
  const float* bq    = (const float*)d_in[2];
  const float* wk    = (const float*)d_in[3];
  const float* bk    = (const float*)d_in[4];
  const float* wv    = (const float*)d_in[5];
  const float* bv    = (const float*)d_in[6];
  const float* gamma = (const float*)d_in[7];
  float* out = (float*)d_out;

  // Workspace layout (~20.7 MB total).
  const size_t QK = (size_t)BATCH * NPIX * C8;  // f16 elems (1 M each)
  _Float16* qT = (_Float16*)d_ws;
  _Float16* kT = qT + QK;
  _Float16* vW = kT + QK;                        // BATCH*CH*NPIX f16
  float* Mst = (float*)(vW + (size_t)BATCH * CH * NPIX);
  float* Lst = Mst + (size_t)BATCH * NPIX;
  _Float16* wqh = (_Float16*)(Lst + (size_t)BATCH * NPIX);
  _Float16* wkh = wqh + (size_t)C8 * CH;
  _Float16* wvh = wkh + (size_t)C8 * CH;

  (void)in_sizes; (void)n_in; (void)out_size; (void)ws_size;

  // 0) weight conversion (once): 65536 elems / 256.
  wcvt_kernel<<<CH * CH / 256, 256, 0, stream>>>(wq, wk, wv, wqh, wkh, wvh);
  // 1) projections: 8 batches x 256 column tiles, 4 waves/block.
  proj_kernel<<<BATCH * NT, 128, 0, stream>>>(input, wqh, bq, wkh, bk, wvh, bv,
                                              qT, kT, vW);
  // 2) softmax stats: 8*256 m-tiles, 8 waves/block.
  stats_kernel<<<BATCH * NT / 8, 256, 0, stream>>>(qT, kT, Mst, Lst);
  // 3) attention output + residual: 8 batches x 64 (64-m) blocks.
  out_kernel<<<BATCH * (NT / 4), 256, 0, stream>>>(
      qT, kT, vW, Mst, Lst, gamma, input, out);
}
